// RoPEMultiHeadAttention_3599182594647
// MI455X (gfx1250) — matrix operand order repaired and checked by reading, not compiled, not run on hardware
//
#include <hip/hip_runtime.h>
#include <hip/hip_bf16.h>

// ---------------------------------------------------------------------------
// Types for CDNA5 WMMA (wave32): v16bf A/B fragments, v8f accumulator.
// ---------------------------------------------------------------------------
typedef __attribute__((ext_vector_type(8)))  float  v8f;
typedef __attribute__((ext_vector_type(8)))  __bf16 v8bf;
typedef __attribute__((ext_vector_type(16))) __bf16 v16bf;

union FragBF {
  v16bf v16;
  v8bf  v8[2];
};

union U16x8 {
  uint4 u;
  unsigned short s[8];
};

union U16x4 {
  uint2 u;
  unsigned short s[4];
};

// float -> bf16 round-to-nearest-even
__device__ __forceinline__ unsigned short f2bf(float f) {
  unsigned int u = __float_as_uint(f);
  u += 0x7FFFu + ((u >> 16) & 1u);
  return (unsigned short)(u >> 16);
}

#define BM 128
#define BN 128
#define BK 32
#define LDSS 48  // LDS row stride in bf16 elems (96B: 16B-aligned rows, skewed banks)

// ---------------------------------------------------------------------------
// Staging helpers. `tile` points at the LDS tile base (generic pointer whose
// low 32 bits are the LDS byte offset, per the gfx1250 aperture layout).
// Each thread covers one 16-elem (32B in LDS) row-chunk: row = tid>>1, col = (tid&1)*16.
// ---------------------------------------------------------------------------
__device__ __forceinline__ void stage_f32(const float* __restrict__ G, int ld,
                                          int rows_valid, int base_row, int k0,
                                          unsigned short* tile, int tid) {
  const int r  = tid >> 1;
  const int cb = (tid & 1) * 16;
  unsigned short* dst = tile + r * LDSS + cb;
  float4 f0, f1, f2, f3;
  const int grow = base_row + r;
  if (grow < rows_valid) {
    const float4* p = (const float4*)(G + (size_t)grow * (size_t)ld + k0 + cb);
    f0 = p[0]; f1 = p[1]; f2 = p[2]; f3 = p[3];
  } else {
    f0 = f1 = f2 = f3 = make_float4(0.f, 0.f, 0.f, 0.f);
  }
  U16x8 lo, hi;
  lo.s[0] = f2bf(f0.x); lo.s[1] = f2bf(f0.y);
  lo.s[2] = f2bf(f0.z); lo.s[3] = f2bf(f0.w);
  lo.s[4] = f2bf(f1.x); lo.s[5] = f2bf(f1.y);
  lo.s[6] = f2bf(f1.z); lo.s[7] = f2bf(f1.w);
  hi.s[0] = f2bf(f2.x); hi.s[1] = f2bf(f2.y);
  hi.s[2] = f2bf(f2.z); hi.s[3] = f2bf(f2.w);
  hi.s[4] = f2bf(f3.x); hi.s[5] = f2bf(f3.y);
  hi.s[6] = f2bf(f3.z); hi.s[7] = f2bf(f3.w);
  *(uint4*)dst       = lo.u;
  *(uint4*)(dst + 8) = hi.u;
}

// bf16 operand: DMA straight into LDS, no VGPR round trip (ASYNCcnt-tracked).
__device__ __forceinline__ void stage_bf16(const unsigned short* __restrict__ G,
                                           int ld, int rows_valid, int base_row,
                                           int k0, unsigned short* tile,
                                           int tid) {
  const int r  = tid >> 1;
  const int cb = (tid & 1) * 16;
  unsigned short* dst = tile + r * LDSS + cb;
  const int grow = base_row + r;
  if (grow < rows_valid) {
    const unsigned short* src = G + (size_t)grow * (size_t)ld + k0 + cb;
    const unsigned ldsoff = (unsigned)(size_t)dst;  // low 32 bits = LDS addr
    asm volatile("global_load_async_to_lds_b128 %0, %1, off" ::"v"(ldsoff),
                 "v"(src)
                 : "memory");
    asm volatile("global_load_async_to_lds_b128 %0, %1, off offset:16" ::"v"(
                     ldsoff),
                 "v"(src)
                 : "memory");
  } else {
    const uint4 z = make_uint4(0u, 0u, 0u, 0u);
    *(uint4*)dst       = z;
    *(uint4*)(dst + 8) = z;
  }
}

// ---------------------------------------------------------------------------
// Generic batched GEMM:  C[M,N] = alpha * A[M,K] * Bt[N,K]^T  (+ bias[N])
// AMODE/BMODE: 0 = f32 operand (convert while staging), 1 = bf16 operand
// (async DMA to LDS).  CMODE: 0 = f32 store, 1 = bf16 store.
// Double-buffered LDS: next K-tile staged (async) while current computes.
// 256 threads = 8 waves; wave grid 4(M) x 2(N); each wave: 2x4 WMMA tiles.
// ---------------------------------------------------------------------------
template <int AMODE, int BMODE, int CMODE>
__global__ __launch_bounds__(256) void gemm_wmma(
    const void* __restrict__ Av, const void* __restrict__ Bv,
    void* __restrict__ Cv, const float* __restrict__ bias, int M, int N, int K,
    int lda, int ldb, int ldc, float alpha, int inner, long long soA,
    long long siA, long long soB, long long siB, long long soC, long long siC,
    int addBias) {
  __shared__ __align__(16) unsigned short As[2][BM * LDSS];
  __shared__ __align__(16) unsigned short Bs[2][BN * LDSS];

  const int tid   = threadIdx.x;
  const int lane  = tid & 31;
  const int wave  = tid >> 5;
  const int waveM = wave >> 1;  // 0..3
  const int waveN = wave & 1;   // 0..1
  const int l15   = lane & 15;
  const int hsel  = lane >> 4;  // lane half-select (ISA fragment layouts)

  const int z  = blockIdx.z;
  const int zo = z / inner;
  const int zi = z - zo * inner;
  const size_t offA = (size_t)zo * (size_t)soA + (size_t)zi * (size_t)siA;
  const size_t offB = (size_t)zo * (size_t)soB + (size_t)zi * (size_t)siB;
  const size_t offC = (size_t)zo * (size_t)soC + (size_t)zi * (size_t)siC;

  const float*          Af  = (const float*)Av + offA;
  const unsigned short* A16 = (const unsigned short*)Av + offA;
  const float*          Bf  = (const float*)Bv + offB;
  const unsigned short* B16 = (const unsigned short*)Bv + offB;

  const int bm = blockIdx.y * BM;
  const int bn = blockIdx.x * BN;

  v8f acc[2][4];
  for (int m = 0; m < 2; ++m)
    for (int n = 0; n < 4; ++n)
      for (int i = 0; i < 8; ++i) acc[m][n][i] = 0.0f;

  auto stageA = [&](int buf, int k0) {
    if (AMODE == 0)
      stage_f32(Af, lda, M, bm, k0, As[buf], tid);
    else
      stage_bf16(A16, lda, M, bm, k0, As[buf], tid);
  };
  auto stageB = [&](int buf, int k0) {
    if (BMODE == 0)
      stage_f32(Bf, ldb, N, bn, k0, Bs[buf], tid);
    else
      stage_bf16(B16, ldb, N, bn, k0, Bs[buf], tid);
  };

  // prologue: stage first tile
  stageA(0, 0);
  stageB(0, 0);
  asm volatile("s_wait_asynccnt 0x0" ::: "memory");
  __syncthreads();

  int p = 0;
  for (int k0 = 0; k0 < K; k0 += BK) {
    const int nk = k0 + BK;
    if (nk < K) {  // kick off next tile's DMA while computing this one
      stageA(1 - p, nk);
      stageB(1 - p, nk);
    }

    // ---- fragment loads per ISA 16-bit layouts (wave32) ----
    // A 16x32: lanes 0-15 row M=l, K {0..7,16..23}; lanes 16-31 K {8..15,24..31}
    FragBF fa[2], fb[4];
#pragma unroll
    for (int m = 0; m < 2; ++m) {
      const int row = waveM * 32 + m * 16 + l15;
      fa[m].v8[0] = *(const v8bf*)&As[p][row * LDSS + hsel * 8];
      fa[m].v8[1] = *(const v8bf*)&As[p][row * LDSS + hsel * 8 + 16];
    }
    // B 32x16: lane n holds column N=n, K {0..7,16..23} (lanes 0-15) / {8..15,24..31} (16-31)
#pragma unroll
    for (int n = 0; n < 4; ++n) {
      const int row = waveN * 64 + n * 16 + l15;
      fb[n].v8[0] = *(const v8bf*)&Bs[p][row * LDSS + hsel * 8];
      fb[n].v8[1] = *(const v8bf*)&Bs[p][row * LDSS + hsel * 8 + 16];
    }
#pragma unroll
    for (int m = 0; m < 2; ++m)
#pragma unroll
      for (int n = 0; n < 4; ++n)
        acc[m][n] = __builtin_amdgcn_wmma_f32_16x16x32_bf16(
            false, fa[m].v16, false, fb[n].v16, (short)0, acc[m][n], false,
            false);

    if (nk < K) {
      asm volatile("s_wait_asynccnt 0x0" ::: "memory");
      __syncthreads();
      p ^= 1;
    }
  }

  // ---- store C per f32 C/D layout: VGPR r -> row (hsel*8 + r), col = l15 ----
#pragma unroll
  for (int m = 0; m < 2; ++m)
#pragma unroll
    for (int n = 0; n < 4; ++n) {
      const int cn = bn + waveN * 64 + n * 16 + l15;
      if (cn < N) {
        const int cm0 = bm + waveM * 32 + m * 16 + hsel * 8;
        const float bv = addBias ? bias[cn] : 0.0f;
#pragma unroll
        for (int r = 0; r < 8; ++r) {
          const float v = acc[m][n][r] * alpha + bv;
          if (CMODE == 0)
            ((float*)Cv)[offC + (size_t)(cm0 + r) * (size_t)ldc + cn] = v;
          else
            ((unsigned short*)Cv)[offC + (size_t)(cm0 + r) * (size_t)ldc + cn] =
                f2bf(v);
        }
      }
    }
}

// ---------------------------------------------------------------------------
// f32 -> bf16 bulk convert (weights), 4 elems/thread.
// ---------------------------------------------------------------------------
__global__ __launch_bounds__(256) void f32_to_bf16(
    const float* __restrict__ s, unsigned short* __restrict__ d, int n) {
  const int i = (blockIdx.x * 256 + threadIdx.x) * 4;
  if (i < n) {
    const float4 f = *(const float4*)(s + i);
    U16x4 u;
    u.s[0] = f2bf(f.x); u.s[1] = f2bf(f.y);
    u.s[2] = f2bf(f.z); u.s[3] = f2bf(f.w);
    *(uint2*)(d + i) = u.u;
  }
}

// ---------------------------------------------------------------------------
// RoPE + head-major repack (outputs bf16). Qp/Kp/Vp are [B*S, D] f32.
//   Qh,Kh: bf16 [B,H,S,dk] rotated;  Vt: bf16 [B,H,dk,S] (for PV GEMM).
// One thread per (b,s,h,i), i = rotation pair 0..31.  B=2,S=2048,H=16,dk=64.
// ---------------------------------------------------------------------------
__global__ __launch_bounds__(256) void rope_pack(
    const float* __restrict__ Qp, const float* __restrict__ Kp,
    const float* __restrict__ Vp, unsigned short* __restrict__ Qh,
    unsigned short* __restrict__ Kh, unsigned short* __restrict__ Vt) {
  const size_t gid = (size_t)blockIdx.x * 256 + threadIdx.x;
  const int i = (int)(gid & 31);  // pair index
  const size_t t = gid >> 5;      // (b*S+s)*H + h
  const int h = (int)(t & 15);
  const size_t bs = t >> 4;       // b*S + s
  const int s = (int)(bs & 2047);
  const int b = (int)(bs >> 11);

  const size_t src = bs * 1024 + (size_t)h * 64;
  const float q1 = Qp[src + i], q2 = Qp[src + i + 32];
  const float k1 = Kp[src + i], k2 = Kp[src + i + 32];

  const float freq = __powf(10000.0f, -(float)(2 * i) / 64.0f);
  float sn, cs;
  __sincosf((float)s * freq, &sn, &cs);

  const size_t dsth = (((size_t)(b * 16 + h)) * 2048 + s) * 64;
  Qh[dsth + i]      = f2bf(q1 * cs - q2 * sn);
  Qh[dsth + i + 32] = f2bf(q2 * cs + q1 * sn);
  Kh[dsth + i]      = f2bf(k1 * cs - k2 * sn);
  Kh[dsth + i + 32] = f2bf(k2 * cs + k1 * sn);

  const size_t dv = (size_t)(b * 16 + h) * 64 * 2048;
  Vt[dv + (size_t)i * 2048 + s]        = f2bf(Vp[src + i]);
  Vt[dv + (size_t)(i + 32) * 2048 + s] = f2bf(Vp[src + i + 32]);
}

// ---------------------------------------------------------------------------
// In-place row softmax over attn: rows = B*H*S = 65536, row length S = 2048.
// ---------------------------------------------------------------------------
__global__ __launch_bounds__(256) void softmax_rows(float* __restrict__ attn) {
  __shared__ float red[256];
  const int tid = threadIdx.x;
  float* p = attn + (size_t)blockIdx.x * 2048;

  float v[8];
  float mx = -3.402823466e38f;
#pragma unroll
  for (int i = 0; i < 8; ++i) {
    v[i] = p[tid + i * 256];
    mx = fmaxf(mx, v[i]);
  }
  red[tid] = mx;
  __syncthreads();
  for (int s = 128; s > 0; s >>= 1) {
    if (tid < s) red[tid] = fmaxf(red[tid], red[tid + s]);
    __syncthreads();
  }
  mx = red[0];
  __syncthreads();

  float sum = 0.0f;
#pragma unroll
  for (int i = 0; i < 8; ++i) {
    v[i] = __expf(v[i] - mx);
    sum += v[i];
  }
  red[tid] = sum;
  __syncthreads();
  for (int s = 128; s > 0; s >>= 1) {
    if (tid < s) red[tid] += red[tid + s];
    __syncthreads();
  }
  const float inv = 1.0f / red[0];
#pragma unroll
  for (int i = 0; i < 8; ++i) p[tid + i * 256] = v[i] * inv;
}

// ---------------------------------------------------------------------------
// Pipeline: weight convert -> 3 projections -> RoPE pack (bf16) -> scores
// GEMM (bf16 async operands, alpha=1/8, into attn out region) -> softmax ->
// PV GEMM (f32 attn convert + bf16 V async, bf16 ctx out) -> out projection.
// ---------------------------------------------------------------------------
extern "C" void kernel_launch(void* const* d_in, const int* in_sizes, int n_in,
                              void* d_out, int out_size, void* d_ws,
                              size_t ws_size, hipStream_t stream) {
  const float* query = (const float*)d_in[0];
  const float* key_  = (const float*)d_in[1];
  const float* value = (const float*)d_in[2];
  const float* Wq    = (const float*)d_in[3];
  const float* Wk    = (const float*)d_in[4];
  const float* Wv    = (const float*)d_in[5];
  const float* Wo    = (const float*)d_in[6];
  const float* bo    = (const float*)d_in[7];

  // B=2,S=2048,D=1024,H=16,dk=64
  float* out  = (float*)d_out;              // [B,S,D]
  float* attn = (float*)d_out + 4194304ll;  // [B,H,S,S]

  float* ws = (float*)d_ws;
  const size_t SZ = 4194304;  // B*S*D elements
  float* Qp = ws;
  float* Kp = ws + SZ;
  float* Vp = ws + 2 * SZ;
  unsigned short* wsb  = (unsigned short*)(ws + 3 * SZ);
  unsigned short* Qh16 = wsb;
  unsigned short* Kh16 = wsb + SZ;
  unsigned short* Vt16 = wsb + 2 * SZ;
  unsigned short* ctx16 = wsb + 3 * SZ;
  unsigned short* Wq16 = wsb + 4 * SZ;
  unsigned short* Wk16 = Wq16 + 1048576;
  unsigned short* Wv16 = Wk16 + 1048576;
  unsigned short* Wo16 = Wv16 + 1048576;

  dim3 blk(256, 1, 1);

  // --- weights f32 -> bf16 (1M elems each, 4/thread) ---
  f32_to_bf16<<<dim3(1024, 1, 1), blk, 0, stream>>>(Wq, Wq16, 1048576);
  f32_to_bf16<<<dim3(1024, 1, 1), blk, 0, stream>>>(Wk, Wk16, 1048576);
  f32_to_bf16<<<dim3(1024, 1, 1), blk, 0, stream>>>(Wv, Wv16, 1048576);
  f32_to_bf16<<<dim3(1024, 1, 1), blk, 0, stream>>>(Wo, Wo16, 1048576);

  // --- input projections: [4096,1024] x [1024,1024]^T (A f32, B bf16) ---
  dim3 gproj(1024 / BN, 4096 / BM, 1);
  gemm_wmma<0, 1, 0><<<gproj, blk, 0, stream>>>(
      query, Wq16, Qp, nullptr, 4096, 1024, 1024, 1024, 1024, 1024, 1.0f, 1, 0,
      0, 0, 0, 0, 0, 0);
  gemm_wmma<0, 1, 0><<<gproj, blk, 0, stream>>>(
      key_, Wk16, Kp, nullptr, 4096, 1024, 1024, 1024, 1024, 1024, 1.0f, 1, 0,
      0, 0, 0, 0, 0, 0);
  gemm_wmma<0, 1, 0><<<gproj, blk, 0, stream>>>(
      value, Wv16, Vp, nullptr, 4096, 1024, 1024, 1024, 1024, 1024, 1.0f, 1, 0,
      0, 0, 0, 0, 0, 0);

  // --- RoPE + repack to bf16 head-major ---
  rope_pack<<<dim3(8192, 1, 1), blk, 0, stream>>>(Qp, Kp, Vp, Qh16, Kh16,
                                                  Vt16);

  // --- scores = (Qh Kh^T)/8 per (b,h): M=N=2048, K=64, batch=32 ---
  {
    const long long soQK = 16ll * 2048 * 64, siQK = 2048ll * 64;
    const long long soS = 16ll * 2048 * 2048, siS = 2048ll * 2048;
    dim3 g(2048 / BN, 2048 / BM, 32);
    gemm_wmma<1, 1, 0><<<g, blk, 0, stream>>>(Qh16, Kh16, attn, nullptr, 2048,
                                              2048, 64, 64, 64, 2048, 0.125f,
                                              16, soQK, siQK, soQK, siQK, soS,
                                              siS, 0);
  }

  // --- softmax rows in place ---
  softmax_rows<<<dim3(65536, 1, 1), blk, 0, stream>>>(attn);

  // --- ctx16 = attn * V: M=2048, N=64, K=2048, batch=32; C -> [b,s,h*64+d] ---
  {
    const long long soS = 16ll * 2048 * 2048, siS = 2048ll * 2048;
    const long long soV = 16ll * 64 * 2048, siV = 64ll * 2048;
    const long long soC = 2048ll * 1024, siC = 64;
    dim3 g(1, 2048 / BM, 32);
    gemm_wmma<0, 1, 1><<<g, blk, 0, stream>>>(attn, Vt16, ctx16, nullptr, 2048,
                                              64, 2048, 2048, 2048, 1024, 1.0f,
                                              16, soS, siS, soV, siV, soC, siC,
                                              0);
  }

  // --- output projection: out = ctx @ Wo^T + bo (both operands bf16) ---
  gemm_wmma<1, 1, 0><<<gproj, blk, 0, stream>>>(
      ctx16, Wo16, out, bo, 4096, 1024, 1024, 1024, 1024, 1024, 1.0f, 1, 0, 0,
      0, 0, 0, 0, 1);

  (void)in_sizes; (void)n_in; (void)out_size; (void)ws_size;
}
